// LPAMLP_72782515798235
// MI455X (gfx1250) — matrix operand order repaired and checked by reading, not compiled, not run on hardware
//
#include <hip/hip_runtime.h>

typedef __attribute__((ext_vector_type(16))) __bf16 v16bf;
typedef __attribute__((ext_vector_type(8)))  __bf16 v8bf;
typedef __attribute__((ext_vector_type(8)))  float  v8f;

#define Bn   4
#define Nn   8192
#define Kn   32
#define Cn   64
#define CINn 67
#define KP   96          // padded inner dim for layer 1 (3 x K=32 WMMA steps)
#define TN   4           // points per block
#define PX   (TN*Kn)     // 128 pixels per block
#define THREADS 256

// ---- CDNA5 WMMA fragment loaders (wave32), contiguous k = 16h+s order in A and B (not the ISA table order) ----
// A 16x32 (MxK) bf16, row-major W[m][k] in LDS:
// lane holds M=lane&15; lanes 0-15: K=0..15, lanes 16-31: K=16..31
// => elements a[0..15] = W[m][kbase+16h .. +16h+15]
__device__ __forceinline__ v16bf lda(const __bf16* W, int mbase, int ld, int kbase, int lane) {
  const int half = lane >> 4;
  const __bf16* r = W + (mbase + (lane & 15)) * ld + kbase + 16 * half;
  const v8bf lo = *(const v8bf*)r;
  const v8bf hi = *(const v8bf*)(r + 8);
  return __builtin_shufflevector(lo, hi, 0, 1, 2, 3, 4, 5, 6, 7,
                                 8, 9, 10, 11, 12, 13, 14, 15);
}

// B 32x16 (KxN) bf16 with X stored pixel-major X[n][k] (row stride ld):
// lane holds N=lane&15; lanes 0-15: K=0..15, lanes 16-31: K=16..31 (VGPR v -> k pair 2v,2v+1)
// => lane's fragment = 16 contiguous bf16 at X[n][kbase + 16*(lane>>4)]
__device__ __forceinline__ v16bf ldb(const __bf16* X, int kbase, int ld, int nbase, int lane) {
  const __bf16* r = X + (nbase + (lane & 15)) * ld + kbase + 16 * (lane >> 4);
  return *(const v16bf*)r;
}

__global__ __launch_bounds__(THREADS)
void lpa_fused(const float* __restrict__ p, const float* __restrict__ f,
               const int* __restrict__ idx,
               const float* __restrict__ Wa,  const float* __restrict__ ba,
               const float* __restrict__ W1,  const float* __restrict__ b1,
               const float* __restrict__ W2,  const float* __restrict__ b2,
               const float* __restrict__ Wf1, const float* __restrict__ bf1,
               const float* __restrict__ Wf2, const float* __restrict__ bf2,
               float* __restrict__ out)
{
  __shared__ __bf16 sWa[Cn * KP];     // row-major [m][k], channels [fj|dp|0]
  __shared__ __bf16 sW1[Cn * KP];
  __shared__ __bf16 sW2[Cn * Cn];     // row-major [m][k]
  __shared__ __bf16 sWf1[Cn * Cn];    // TRANSPOSED [j][c]
  __shared__ __bf16 sWf2[Cn * Cn];    // TRANSPOSED [j][c]
  __shared__ __bf16 sX[PX * KP];      // pixel-major [pixel][ch]; reused for xi
  __shared__ __bf16 sH[PX * Cn];      // pixel-major
  __shared__ __bf16 sA[PX * Cn];      // pixel-major
  __shared__ float  sFo[Cn * TN];
  __shared__ float  sG[Cn * TN];
  __shared__ float  sBias[5 * Cn];

  const int tid = threadIdx.x;
  const int b   = blockIdx.x / (Nn / TN);
  const int n0  = (blockIdx.x % (Nn / TN)) * TN;

  // ---- stage weights (bf16; layer-1 channels reordered [fj(64)|dp(3)|zero(29)]) ----
  for (int e = tid; e < Cn * KP; e += THREADS) {
    const int o = e / KP, j = e % KP;
    float wa = 0.f, w1 = 0.f;
    if (j < 64)      { wa = Wa[o * CINn + 3 + j];    w1 = W1[o * CINn + 3 + j]; }
    else if (j < 67) { wa = Wa[o * CINn + (j - 64)]; w1 = W1[o * CINn + (j - 64)]; }
    sWa[e] = (__bf16)wa;
    sW1[e] = (__bf16)w1;
  }
  for (int e = tid; e < Cn * Cn; e += THREADS) {
    const int c = e >> 6, j = e & 63;
    sW2[e] = (__bf16)W2[e];
    sWf1[j * Cn + c] = (__bf16)Wf1[e];   // transpose
    sWf2[j * Cn + c] = (__bf16)Wf2[e];   // transpose
  }
  if (tid < Cn) {
    sBias[tid]          = ba[tid];
    sBias[Cn + tid]     = b1[tid];
    sBias[2 * Cn + tid] = b2[tid];
    sBias[3 * Cn + tid] = bf1[tid];
    sBias[4 * Cn + tid] = bf2[tid];
  }

  // ---- gather: build X [128 pixels][96 ch] in LDS (pixel-major, vector stores) ----
  {
    const int px  = tid & (PX - 1);
    const int grp = tid >> 7;                  // 0..1
    const int pt  = px >> 5, k = px & (Kn - 1);
    const int n   = n0 + pt;
    const int j   = idx[(b * Nn + n) * Kn + k];
#pragma unroll
    for (int q = 0; q < 4; ++q) {              // 32 contiguous channels per thread
      v8bf t;
#pragma unroll
      for (int e = 0; e < 8; ++e)
        t[e] = (__bf16)f[(b * Cn + grp * 32 + q * 8 + e) * Nn + j];
      *(v8bf*)&sX[px * KP + grp * 32 + q * 8] = t;
    }
    if (grp == 0) {                            // channels 64..95: dp(3) + zeros(29)
      v8bf t0;
#pragma unroll
      for (int d = 0; d < 3; ++d)
        t0[d] = (__bf16)(p[(b * Nn + j) * 3 + d] - p[(b * Nn + n) * 3 + d]);
#pragma unroll
      for (int e = 3; e < 8; ++e) t0[e] = (__bf16)0.f;
      *(v8bf*)&sX[px * KP + 64] = t0;
      v8bf z;
#pragma unroll
      for (int e = 0; e < 8; ++e) z[e] = (__bf16)0.f;
#pragma unroll
      for (int q = 0; q < 3; ++q)
        *(v8bf*)&sX[px * KP + 72 + q * 8] = z;
    }
  }
  __syncthreads();

  const int lane = tid & 31;
  const int wave = tid >> 5;
  const int nb   = wave * 16;          // each wave owns one 16-pixel tile
  const int half = lane >> 4;
  const int ncol = nb + (lane & 15);

  // ---- attn & h GEMMs: [64x96] x [96x16] per wave (B fragment shared) ----
#pragma unroll
  for (int mt = 0; mt < 4; ++mt) {
    v8f acc  = {};
    v8f acch = {};
#pragma unroll
    for (int ks = 0; ks < 3; ++ks) {
      const v16bf bb = ldb(sX, ks * 32, KP, nb, lane);
      const v16bf aa = lda(sWa, mt * 16, KP, ks * 32, lane);
      acc = __builtin_amdgcn_wmma_f32_16x16x32_bf16(false, aa, false, bb,
                                                    (short)0, acc, false, false);
      const v16bf a1 = lda(sW1, mt * 16, KP, ks * 32, lane);
      acch = __builtin_amdgcn_wmma_f32_16x16x32_bf16(false, a1, false, bb,
                                                     (short)0, acch, false, false);
    }
    const int m0 = mt * 16 + 8 * half;  // 8 consecutive output channels per lane
    v8bf oa, oh;
#pragma unroll
    for (int r = 0; r < 8; ++r) {
      oa[r] = (__bf16)(acc[r] + sBias[m0 + r]);
      const float hv = acch[r] + sBias[Cn + m0 + r];
      oh[r] = (__bf16)(hv > 0.f ? hv : 0.f);
    }
    *(v8bf*)&sA[ncol * Cn + m0] = oa;   // one ds_store_b128 each
    *(v8bf*)&sH[ncol * Cn + m0] = oh;
  }
  __syncthreads();

  // ---- xi GEMM: [64x64] x [64x16]; result overwrites sX (stride KP) ----
#pragma unroll
  for (int mt = 0; mt < 4; ++mt) {
    v8f acc = {};
#pragma unroll
    for (int ks = 0; ks < 2; ++ks) {
      const v16bf aa = lda(sW2, mt * 16, Cn, ks * 32, lane);
      const v16bf bb = ldb(sH, ks * 32, Cn, nb, lane);
      acc = __builtin_amdgcn_wmma_f32_16x16x32_bf16(false, aa, false, bb,
                                                    (short)0, acc, false, false);
    }
    const int m0 = mt * 16 + 8 * half;
    v8bf ox;
#pragma unroll
    for (int r = 0; r < 8; ++r)
      ox[r] = (__bf16)(acc[r] + sBias[2 * Cn + m0 + r]);
    *(v8bf*)&sX[ncol * KP + m0] = ox;
  }
  __syncthreads();

  // ---- softmax over K + weighted sum + residual + relu (one (c,pt) per thread) ----
  // lanes sweep consecutive channels -> conflict-free LDS reads
  {
    const int c  = tid & (Cn - 1);
    const int pt = tid >> 6;
    float mx = -1e30f;
    for (int k = 0; k < Kn; ++k) {
      const float a = (float)sA[(pt * Kn + k) * Cn + c];
      mx = a > mx ? a : mx;
    }
    float s = 0.f, fo = 0.f;
    for (int k = 0; k < Kn; ++k) {
      const float e = __expf((float)sA[(pt * Kn + k) * Cn + c] - mx);
      s  += e;
      fo += e * (float)sX[(pt * Kn + k) * KP + c];      // xi
    }
    fo /= s;
    fo += f[(b * Cn + c) * Nn + (n0 + pt)];             // residual
    sFo[c * TN + pt] = fo > 0.f ? fo : 0.f;
  }
  __syncthreads();

  // ---- FFN layer 1 (tiny; scalar, conflict-free via transposed weights) ----
  {
    const int c  = tid & (Cn - 1);
    const int pt = tid >> 6;
    float g = sBias[3 * Cn + c];
    for (int j = 0; j < Cn; ++j)
      g += (float)sWf1[j * Cn + c] * sFo[j * TN + pt];
    sG[c * TN + pt] = g > 0.f ? g : 0.f;
  }
  __syncthreads();

  // ---- FFN layer 2 + residual + relu -> output ----
  {
    const int c  = tid & (Cn - 1);
    const int pt = tid >> 6;
    float g = sBias[4 * Cn + c];
    for (int j = 0; j < Cn; ++j)
      g += (float)sWf2[j * Cn + c] * sG[j * TN + pt];
    float f2 = g + sFo[c * TN + pt];
    f2 = f2 > 0.f ? f2 : 0.f;
    out[Bn * Nn * 3 + (b * Cn + c) * Nn + (n0 + pt)] = f2;
  }
}

__global__ void copy_p_kernel(const float* __restrict__ p, float* __restrict__ out) {
  const int i = blockIdx.x * blockDim.x + threadIdx.x;
  if (i < Bn * Nn * 3) out[i] = p[i];
}

extern "C" void kernel_launch(void* const* d_in, const int* in_sizes, int n_in,
                              void* d_out, int out_size, void* d_ws, size_t ws_size,
                              hipStream_t stream) {
  (void)in_sizes; (void)n_in; (void)out_size; (void)d_ws; (void)ws_size;
  const float* p   = (const float*)d_in[0];
  const float* f   = (const float*)d_in[1];
  const int*   idx = (const int*)d_in[2];
  const float* Wa  = (const float*)d_in[3];
  const float* ba  = (const float*)d_in[4];
  const float* W1  = (const float*)d_in[5];
  const float* b1  = (const float*)d_in[6];
  const float* W2  = (const float*)d_in[7];
  const float* b2  = (const float*)d_in[8];
  const float* Wf1 = (const float*)d_in[9];
  const float* bf1 = (const float*)d_in[10];
  const float* Wf2 = (const float*)d_in[11];
  const float* bf2 = (const float*)d_in[12];
  float* out = (float*)d_out;

  copy_p_kernel<<<(Bn * Nn * 3 + 255) / 256, 256, 0, stream>>>(p, out);
  lpa_fused<<<Bn * (Nn / TN), THREADS, 0, stream>>>(
      p, f, idx, Wa, ba, W1, b1, W2, b2, Wf1, bf1, Wf2, bf2, out);
}
